// ConditionalGRU_532575944814
// MI455X (gfx1250) — compile-verified, matrix operand order checked by reading, not run on hardware
//
#include <hip/hip_runtime.h>
#include <hip/hip_bf16.h>

// ---------------------------------------------------------------------------
// ConditionalGRU for MI455X (gfx1250, wave32, WMMA bf16 -> f32 accumulate)
//   B=64, T=256, D=512, H=1024
//
// v2: single PERSISTENT kernel for the 256-step recurrence.
//   * 64 blocks x 256 threads (8 waves). Each block owns a fixed slice of the
//     gate columns and preloads its weights into LDS ONCE:
//       - 32 KA columns (g = blk*32..+31):  Wi[g][0:512] ++ Wh[g][0:1024]  (96KB)
//       - 16 KB columns (g = blk*16..+15):  Wc[g][0:1024] ++ Wi[2H+g][0:512] (48KB)
//     ~145KB of LDS -- only possible because CDNA5 WGPs have 320KB LDS.
//     Column stride padded to 1544 elems so ds_load_b128 lanes hit distinct
//     bank groups.
//   * Per step: phase1 (8 waves: r,i gates -> rh bf16 + inputgate), device-wide
//     split barrier (atomic arrive + s_sleep poll), phase2 (4 waves: n gate,
//     tanh, state update, y[b,t,:]), barrier. 2 barriers/step instead of 2
//     kernel launches/step -- launch overhead was the dominant serial cost.
//   * All GEMMs are X*W^T with row-major W => A rows and B columns contiguous
//     along K => every WMMA fragment is exactly two b128 loads per lane
//     (global for activations, ds for weights), matching the CDNA5 16-bit
//     fragment layout: lane-half hh holds K in [8hh,8hh+8) u [16+8hh,+8).
// ---------------------------------------------------------------------------

#define B_ 64
#define T_ 256
#define D_ 512
#define H_ 1024

#define NBLK 64
#define LDS_STRIDE 1544                       // padded col stride (bf16 elems)
#define LDS_COLS   48                         // 32 KA cols + 16 KB cols
#define LDS_BYTES  (LDS_COLS * LDS_STRIDE * 2)   // 148224 B < 320KB WGP LDS

typedef __attribute__((ext_vector_type(16))) __bf16 v16bf;
typedef __attribute__((ext_vector_type(8)))  __bf16 v8bf;
typedef __attribute__((ext_vector_type(8)))  float  v8f;

// round-to-nearest-even fp32 -> bf16 (bit manipulation: always compiles)
__device__ __forceinline__ __bf16 f2bf(float f) {
    unsigned u = __builtin_bit_cast(unsigned, f);
    u = u + 0x7FFFu + ((u >> 16) & 1u);
    unsigned short s = (unsigned short)(u >> 16);
    return __builtin_bit_cast(__bf16, s);
}

// One A (16x32) / B (32x16) bf16 fragment slice for this lane; base points at
// this lane's row/column at K=0, data contiguous along K. Works for both
// global and LDS pointers (addrspace inferred -> global_load_b128/ds_load_b128).
__device__ __forceinline__ v16bf load_frag(const __bf16* base, int k0, int hh) {
    const __bf16* p = base + k0 + 8 * hh;
    v8bf lo = *reinterpret_cast<const v8bf*>(p);        // K = k0+8h .. +7
    v8bf hi = *reinterpret_cast<const v8bf*>(p + 16);   // K = k0+16+8h .. +7
    return __builtin_shufflevector(lo, hi, 0, 1, 2, 3, 4, 5, 6, 7,
                                           8, 9, 10, 11, 12, 13, 14, 15);
}

__device__ __forceinline__ v8f wmma_bf16(v16bf a, v16bf b, v8f c) {
    return __builtin_amdgcn_wmma_f32_16x16x32_bf16(
        false, a, false, b, (short)0, c, false, false);
}

__device__ __forceinline__ float sigmoidf_(float x) {
    return 1.0f / (1.0f + __expf(-x));
}

// Device-wide split barrier: monotonic counter, release stores before arrive,
// acquire after completion; s_sleep while polling. 64 blocks << residency.
__device__ __forceinline__ void grid_barrier(unsigned* ctr, unsigned target) {
    __threadfence();                     // make my stores device-visible
    __syncthreads();
    if (threadIdx.x == 0) {
        __hip_atomic_fetch_add(ctr, 1u, __ATOMIC_RELEASE,
                               __HIP_MEMORY_SCOPE_AGENT);
        unsigned v;
        do {
            __builtin_amdgcn_s_sleep(2);
            v = __hip_atomic_load(ctr, __ATOMIC_ACQUIRE,
                                  __HIP_MEMORY_SCOPE_AGENT);
        } while (v < target);
    }
    __syncthreads();
    __threadfence();                     // drop stale lines before re-reading
}

// ---------------------------------------------------------------------------
// Prologue: conversions + h init
// ---------------------------------------------------------------------------
__global__ __launch_bounds__(256) void cvt_f32_bf16(const float* __restrict__ src,
                                                    __bf16* __restrict__ dst, int n) {
    int i = blockIdx.x * 256 + threadIdx.x;
    if (i < n) dst[i] = f2bf(src[i]);
}

__global__ __launch_bounds__(256) void init_h(const float* __restrict__ h0,
                                              float* __restrict__ h,
                                              __bf16* __restrict__ hb, int n) {
    int i = blockIdx.x * 256 + threadIdx.x;
    if (i < n) { float v = h0[i]; h[i] = v; hb[i] = f2bf(v); }
}

// ---------------------------------------------------------------------------
// Peephole: p[b, 0:3H] = ctx @ Wp^T + bp    (M=64, N=3072, K=1024), once.
// ---------------------------------------------------------------------------
__global__ __launch_bounds__(256) void peephole_gemm(
    const __bf16* __restrict__ ctxb, const __bf16* __restrict__ Wpb,
    const float* __restrict__ bp, float* __restrict__ p) {
    int wave = (blockIdx.x * 256 + threadIdx.x) >> 5;
    int lane = threadIdx.x & 31;
    int hh = lane >> 4, lm = lane & 15;
    int m0 = (wave & 3) * 16;
    int n0 = (wave >> 2) * 16;

    const __bf16* A  = ctxb + (size_t)(m0 + lm) * H_;
    const __bf16* Bp = Wpb + (size_t)(n0 + lm) * H_;

    v8f acc = {};
#pragma unroll 8
    for (int k = 0; k < H_; k += 32)
        acc = wmma_bf16(load_frag(A, k, hh), load_frag(Bp, k, hh), acc);

    int g = n0 + lm;
#pragma unroll
    for (int v = 0; v < 8; ++v) {
        int b = m0 + v + 8 * hh;
        p[(size_t)b * (3 * H_) + g] = acc[v] + bp[g];
    }
}

// ---------------------------------------------------------------------------
// Persistent recurrence kernel: all 256 steps, weights LDS-resident.
// ---------------------------------------------------------------------------
extern __shared__ __align__(16) __bf16 lds_w[];

__global__ __launch_bounds__(256) void gru_persistent(
    const __bf16* __restrict__ xb, __bf16* __restrict__ hb,
    const __bf16* __restrict__ Wib, const __bf16* __restrict__ Whb,
    const __bf16* __restrict__ Wcb,
    const float* __restrict__ bi, const float* __restrict__ bh,
    const float* __restrict__ bc, const float* __restrict__ p,
    float* __restrict__ h, float* __restrict__ inputgate,
    __bf16* __restrict__ rhb, float* __restrict__ out,
    float* __restrict__ hT, unsigned* __restrict__ ctr) {
    const int blk  = blockIdx.x;
    const int w    = threadIdx.x >> 5;          // wave 0..7
    const int lane = threadIdx.x & 31;
    const int hh   = lane >> 4;
    const int lm   = lane & 15;

    // ---- preload this block's weight slice into LDS (once, b128 chunks) ----
    // KA cols c in [0,32): g = blk*32 + c, K = Wi[g][0:512] ++ Wh[g][0:1024]
    for (int e = threadIdx.x; e < 32 * 192; e += 256) {
        int c = e / 192, kc = (e % 192) * 8;
        int g = blk * 32 + c;
        const __bf16* src = (kc < D_) ? (Wib + (size_t)g * D_ + kc)
                                      : (Whb + (size_t)g * H_ + (kc - D_));
        *reinterpret_cast<v8bf*>(lds_w + (size_t)c * LDS_STRIDE + kc) =
            *reinterpret_cast<const v8bf*>(src);
    }
    // KB cols c in [0,16): g = blk*16 + c, K = Wc[g][0:1024] ++ Wi[2H+g][0:512]
    for (int e = threadIdx.x; e < 16 * 192; e += 256) {
        int c = e / 192, kc = (e % 192) * 8;
        int g = blk * 16 + c;
        const __bf16* src = (kc < H_)
            ? (Wcb + (size_t)g * H_ + kc)
            : (Wib + (size_t)(2 * H_ + g) * D_ + (kc - H_));
        *reinterpret_cast<v8bf*>(lds_w + (size_t)(32 + c) * LDS_STRIDE + kc) =
            *reinterpret_cast<const v8bf*>(src);
    }
    __syncthreads();

    unsigned epoch = 0;
    for (int t = 0; t < T_; ++t) {
        // ================= phase 1: r,i gates (8 waves, 1 tile each) ========
        {
            int m0    = (w & 3) * 16;
            int cbase = (w >> 2) * 16;              // 0 or 16
            int g     = blk * 32 + cbase + lm;      // gate column
            const __bf16* Bl = lds_w + (size_t)(cbase + lm) * LDS_STRIDE;
            const __bf16* Ax = xb + ((size_t)(m0 + lm) * T_ + t) * D_;
            const __bf16* Ah = hb + (size_t)(m0 + lm) * H_;

            v8f acc = {};
#pragma unroll 8
            for (int k = 0; k < D_; k += 32)        // x_t @ Wi[g]^T
                acc = wmma_bf16(load_frag(Ax, k, hh), load_frag(Bl, k, hh), acc);
#pragma unroll 8
            for (int k = 0; k < H_; k += 32)        // h @ Wh[g]^T
                acc = wmma_bf16(load_frag(Ah, k, hh),
                                load_frag(Bl, D_ + k, hh), acc);

            float big = bi[g] + bh[g];
#pragma unroll
            for (int v = 0; v < 8; ++v) {
                int b = m0 + v + 8 * hh;
                float s = sigmoidf_(acc[v] + big + p[(size_t)b * (3 * H_) + g]);
                if (g < H_) {       // uniform per tile
                    rhb[(size_t)b * H_ + g] = f2bf(s * h[(size_t)b * H_ + g]);
                } else {
                    inputgate[(size_t)b * H_ + (g - H_)] = s;
                }
            }
        }
        grid_barrier(ctr, NBLK * (++epoch));

        // ================= phase 2: n gate + update (waves 0..3) ============
        if (w < 4) {
            int m0 = w * 16;
            int g  = blk * 16 + lm;
            const __bf16* Bl = lds_w + (size_t)(32 + lm) * LDS_STRIDE;
            const __bf16* Ar = rhb + (size_t)(m0 + lm) * H_;
            const __bf16* Ax = xb + ((size_t)(m0 + lm) * T_ + t) * D_;

            v8f acc = {};
#pragma unroll 8
            for (int k = 0; k < H_; k += 32)        // (r*h) @ Wc[g]^T
                acc = wmma_bf16(load_frag(Ar, k, hh), load_frag(Bl, k, hh), acc);
#pragma unroll 8
            for (int k = 0; k < D_; k += 32)        // x_t @ Wi[2H+g]^T
                acc = wmma_bf16(load_frag(Ax, k, hh),
                                load_frag(Bl, H_ + k, hh), acc);

            float bias = bc[g] + bi[2 * H_ + g];
#pragma unroll
            for (int v = 0; v < 8; ++v) {
                int b = m0 + v + 8 * hh;
                float pre = acc[v] + bias + p[(size_t)b * (3 * H_) + 2 * H_ + g];
                float n = tanhf(pre);
                size_t ix = (size_t)b * H_ + g;
                float hv = h[ix];
                float hy = hv + inputgate[ix] * (n - hv);
                h[ix]  = hy;
                hb[ix] = f2bf(hy);
                out[(size_t)b * T_ * H_ + (size_t)t * H_ + g] = hy;
                if (t == T_ - 1) hT[ix] = hy;
            }
        }
        grid_barrier(ctr, NBLK * (++epoch));
    }
}

// ---------------------------------------------------------------------------
extern "C" void kernel_launch(void* const* d_in, const int* in_sizes, int n_in,
                              void* d_out, int out_size, void* d_ws, size_t ws_size,
                              hipStream_t stream) {
    const float* x   = (const float*)d_in[0];
    const float* h0  = (const float*)d_in[1];
    const float* ctx = (const float*)d_in[2];
    const float* Wi  = (const float*)d_in[3];
    const float* bi  = (const float*)d_in[4];
    const float* Wh  = (const float*)d_in[5];
    const float* bh  = (const float*)d_in[6];
    const float* Wp  = (const float*)d_in[7];
    const float* bp  = (const float*)d_in[8];
    const float* Wc  = (const float*)d_in[9];
    const float* bc  = (const float*)d_in[10];

    float* out = (float*)d_out;
    float* hT  = out + (size_t)B_ * T_ * H_;

    // workspace carve-up (~34 MB total)
    char* ws = (char*)d_ws;
    size_t off = 0;
    auto alloc = [&](size_t bytes) -> char* {
        char* ptr = ws + off;
        off += (bytes + 255) & ~(size_t)255;
        return ptr;
    };
    unsigned* ctr  = (unsigned*)alloc(256);
    __bf16* xb   = (__bf16*)alloc((size_t)B_ * T_ * D_ * 2);
    __bf16* Wib  = (__bf16*)alloc((size_t)3 * H_ * D_ * 2);
    __bf16* Whb  = (__bf16*)alloc((size_t)2 * H_ * H_ * 2);
    __bf16* Wcb  = (__bf16*)alloc((size_t)H_ * H_ * 2);
    __bf16* Wpb  = (__bf16*)alloc((size_t)3 * H_ * H_ * 2);
    __bf16* ctxb = (__bf16*)alloc((size_t)B_ * H_ * 2);
    float*  p    = (float*)alloc((size_t)B_ * 3 * H_ * 4);
    float*  h    = (float*)alloc((size_t)B_ * H_ * 4);
    __bf16* hb   = (__bf16*)alloc((size_t)B_ * H_ * 2);
    __bf16* rhb  = (__bf16*)alloc((size_t)B_ * H_ * 2);
    float*  ig   = (float*)alloc((size_t)B_ * H_ * 4);

    // barrier counter must start at 0 every call (graph-capture safe)
    hipMemsetAsync(ctr, 0, sizeof(unsigned), stream);

    // ---- prologue: bf16 conversions (weights then live in L2 for preload)
    auto cvt = [&](const float* s, __bf16* d, int n) {
        cvt_f32_bf16<<<(n + 255) / 256, 256, 0, stream>>>(s, d, n);
    };
    cvt(x,   xb,   B_ * T_ * D_);
    cvt(Wi,  Wib,  3 * H_ * D_);
    cvt(Wh,  Whb,  2 * H_ * H_);
    cvt(Wc,  Wcb,  H_ * H_);
    cvt(Wp,  Wpb,  3 * H_ * H_);
    cvt(ctx, ctxb, B_ * H_);
    init_h<<<(B_ * H_ + 255) / 256, 256, 0, stream>>>(h0, h, hb, B_ * H_);

    // ---- peephole (once): p = ctx @ Wp^T + bp
    peephole_gemm<<<96, 256, 0, stream>>>(ctxb, Wpb, bp, p);

    // ---- persistent recurrence: 256 steps, 2 device barriers/step
    gru_persistent<<<NBLK, 256, LDS_BYTES, stream>>>(
        xb, hb, Wib, Whb, Wcb, bi, bh, bc, p, h, ig, rhb, out, hT, ctr);
}
